// VQVAE_21397527069380
// MI455X (gfx1250) — matrix operand order repaired and checked by reading, not compiled, not run on hardware
//
#include <hip/hip_runtime.h>
#include <hip/hip_bf16.h>

typedef __attribute__((ext_vector_type(16))) __bf16 v16bf;
typedef __attribute__((ext_vector_type(8)))  __bf16 v8bf;
typedef __attribute__((ext_vector_type(8)))  float  v8f;

#define KC 32            // K chunk per WMMA (bf16 k=32)
#define LRELU_SLOPE 0.01f
#define BN_EPS 1e-5f

__device__ __forceinline__ unsigned short f2bf(float f) {
    unsigned int x = __builtin_bit_cast(unsigned int, f);
    unsigned int r = x + 0x7FFFu + ((x >> 16) & 1u);   // round-to-nearest-even
    return (unsigned short)(r >> 16);
}

__device__ __forceinline__ v16bf ld_frag(const unsigned short* p0, const unsigned short* p1) {
    v8bf lo = *(const v8bf*)p0;
    v8bf hi = *(const v8bf*)p1;
    return __builtin_shufflevector(lo, hi, 0,1,2,3,4,5,6,7,8,9,10,11,12,13,14,15);
}

// ---------------------------------------------------------------------------
// Implicit-GEMM 3x3 SAME conv, bf16 WMMA, f32 accumulate.
// GEMM: M = Cout, N = B*H*W, K = Cin*9. W = H = power of two for every layer,
// so pixel decomposition is shift/mask and a 128-pixel tile stays in-image.
// Block: 256 thr = 8 waves; tile TM Cout x 128 pixels.
// Wave: (TM/4) Cout x 64 px -> AF = TM/64 A-frags x 4 B-frags = 4*AF WMMAs
// per K-chunk, one A-frag set amortized over all B-frags.
// UP=1 fuses nearest-2x upsample into the im2col read (input is H/2 x W/2).
// ---------------------------------------------------------------------------
template<int UP, int TM>
__global__ __launch_bounds__(256) void conv3x3_wmma(
    const float* __restrict__ in, const float* __restrict__ wgt,
    const float* __restrict__ bias, float* __restrict__ out,
    int Cin, int H, int W, int Cout, int lw /*log2(W)*/)
{
    constexpr int AF = TM / 64;               // A fragments per wave (1 or 2)
    __shared__ unsigned short aS[TM * KC];    // [co_loc][k]
    __shared__ unsigned short bS[128 * KC];   // [pix_loc][k] (pixel-major)

    const int HW   = H * W;
    const int K    = Cin * 9;
    const int pixBase = blockIdx.x * 128;
    const int coBase  = blockIdx.y * TM;
    const int tid  = threadIdx.x;
    const int lane = tid & 31;
    const int wave = tid >> 5;
    const int wm   = wave & 3;     // cout sub-tile 0..3 (each AF*16 wide)
    const int wn   = wave >> 2;    // pixel half 0..1

    // Uniform per-block pixel decomposition (only divides in the kernel).
    const int bIdx = pixBase / HW;
    const int rem0 = pixBase - bIdx * HW;

    // Per-thread staging coords: k position is loop-invariant.
    const int kB  = tid & 31;
    const int sub = tid >> 5;                 // 0..7
    int ci = kB / 9;
    int rs = kB - ci * 9;

    const int wstep = Cin * 9;

    v8f acc[AF][4] = {};

    for (int k0 = 0; k0 < K; k0 += KC) {
        const bool vK = (k0 + kB) < K;

        // ---- stage A (weights): TM x 32 ----
        {
            int base = (coBase * Cin + ci) * 9 + rs;
            #pragma unroll
            for (int i = 0; i < TM / 8; ++i) {
                int co   = (i << 3) + sub;
                bool val = vK && (coBase + co) < Cout;
                int addr = val ? (base + co * wstep) : 0;
                float v  = wgt[addr];
                v        = val ? v : 0.f;
                aS[co * KC + kB] = f2bf(v);
            }
        }
        // ---- stage B (im2col): 128 px x 32 k, pixel-major ----
        {
            int d3 = (rs >= 6) ? 2 : ((rs >= 3) ? 1 : 0);
            int dx = rs - d3 * 3 - 1;
            int dy = d3 - 1;
            int cb0 = UP ? (bIdx * Cin + ci) * (HW >> 2)
                         : (bIdx * Cin + ci) * HW;
            #pragma unroll
            for (int i = 0; i < 16; ++i) {
                int pl = (i << 3) + sub;          // 0..127
                int t  = rem0 + pl;
                int y  = t >> lw;
                int x  = t & (W - 1);
                int iy = y + dy, ix = x + dx;
                bool val = vK && ((unsigned)iy < (unsigned)H)
                              && ((unsigned)ix < (unsigned)W);
                int addr;
                if (UP) addr = cb0 + ((iy >> 1) << (lw - 1)) + (ix >> 1);
                else    addr = cb0 + (iy << lw) + ix;
                addr = val ? addr : 0;
                float v = in[addr];
                v = val ? v : 0.f;
                bS[pl * KC + kB] = f2bf(v);
            }
        }
        __syncthreads();

        // ---- fragments (CDNA5 16-bit layouts) + WMMAs ----
        const int row = lane & 15;
        const int hiA = (lane >= 16) ? 8  : 0;
        const int hiB = (lane >= 16) ? 8 : 0;

        v16bf a[AF];
        #pragma unroll
        for (int af = 0; af < AF; ++af) {
            const unsigned short* ar = &aS[((wm * AF + af) * 16 + row) * KC];
            a[af] = ld_frag(ar + hiA, ar + 16 + hiA);
        }

        #pragma unroll
        for (int j = 0; j < 4; ++j) {
            const unsigned short* br = &bS[(wn * 64 + j * 16 + row) * KC + hiB];
            v16bf b = ld_frag(br, br + 16);
            #pragma unroll
            for (int af = 0; af < AF; ++af)
                acc[af][j] = __builtin_amdgcn_wmma_f32_16x16x32_bf16(
                                 false, a[af], false, b, (short)0,
                                 acc[af][j], false, false);
        }
        __syncthreads();

        // advance (ci, rs) by 32 = 3*9 + 5
        ci += 3; rs += 5;
        if (rs >= 9) { rs -= 9; ci += 1; }
    }

    // ---- epilogue: C/D layout VGPR r -> M = r (+8 lanes 16-31), N = lane&15
    const int row = lane & 15;
    const int mo  = (lane >= 16) ? 8 : 0;
    #pragma unroll
    for (int af = 0; af < AF; ++af) {
        #pragma unroll
        for (int r = 0; r < 8; ++r) {
            int co = coBase + (wm * AF + af) * 16 + r + mo;
            if (co >= Cout) continue;
            float bv = bias[co];
            int obase = (bIdx * Cout + co) * HW + rem0;
            #pragma unroll
            for (int j = 0; j < 4; ++j) {
                int off = wn * 64 + j * 16 + row;   // < 128, tile is in-image
                out[obase + off] = acc[af][j][r] + bv;
            }
        }
    }
}

// ---------------------------------------------------------------------------
// BatchNorm batch statistics: one block per channel, deterministic tree reduce.
// ---------------------------------------------------------------------------
__global__ __launch_bounds__(256) void bn_stats_k(
    const float* __restrict__ x, float* __restrict__ stats,
    int C, int HW, int B)
{
    const int c = blockIdx.x;
    const int Nper = B * HW;
    __shared__ float s1[256], s2[256];
    float sum = 0.f, sq = 0.f;
    for (int b = 0; b < B; ++b) {
        const float* p = x + (long long)(b * C + c) * HW;
        for (int i = threadIdx.x; i < HW; i += 256) {
            float v = p[i];
            sum += v; sq += v * v;
        }
    }
    s1[threadIdx.x] = sum; s2[threadIdx.x] = sq;
    __syncthreads();
    for (int s = 128; s > 0; s >>= 1) {
        if (threadIdx.x < s) {
            s1[threadIdx.x] += s1[threadIdx.x + s];
            s2[threadIdx.x] += s2[threadIdx.x + s];
        }
        __syncthreads();
    }
    if (threadIdx.x == 0) {
        float m = s1[0] / (float)Nper;
        stats[c]     = m;
        stats[C + c] = s2[0] / (float)Nper - m * m;
    }
}

// C and HW are powers of two -> shift/mask channel indexing.
__global__ void bn_lrelu_k(float* __restrict__ x, const float* __restrict__ stats,
                           const float* __restrict__ g, const float* __restrict__ bb,
                           int Cmask, int lhw, int total)
{
    int i = blockIdx.x * blockDim.x + threadIdx.x;
    if (i >= total) return;
    int c = (i >> lhw) & Cmask;
    float y = (x[i] - stats[c]) * rsqrtf(stats[Cmask + 1 + c] + BN_EPS) * g[c] + bb[c];
    x[i] = (y > 0.f) ? y : LRELU_SLOPE * y;
}

// H = W = 1<<lw (power of two).
__global__ void maxpool2_k(const float* __restrict__ in, float* __restrict__ out,
                           int total, int lw)
{
    int i = blockIdx.x * blockDim.x + threadIdx.x;
    if (i >= total) return;
    const int W = 1 << lw, Wo = W >> 1;
    int x  = i & (Wo - 1);
    int t  = i >> (lw - 1);
    int y  = t & (Wo - 1);
    int bc = t >> (lw - 1);
    const float* p = in + ((long long)(bc << lw) + (y << 1)) * W + (x << 1);
    out[i] = fmaxf(fmaxf(p[0], p[1]), fmaxf(p[W], p[W + 1]));
}

__global__ void conv1x1_k(const float* __restrict__ in, const float* __restrict__ w,
                          const float* __restrict__ bias, float* __restrict__ out,
                          int Cin, int Cout, int total /*B*Cout*1024*/)
{
    int i = blockIdx.x * blockDim.x + threadIdx.x;
    if (i >= total) return;
    int pix = i & 1023; int t = i >> 10; int co = t % 5; int b = t / 5;
    float s = bias[co];
    const float* ip = in + (long long)b * Cin * 1024 + pix;
    const float* wp = w + (long long)co * Cin;
    for (int c2 = 0; c2 < Cin; ++c2) s += ip[(long long)c2 << 10] * wp[c2];
    out[i] = s;
}

__global__ void vq_k(const float* __restrict__ ze, const float* __restrict__ cb,
                     float* __restrict__ zq, int total /*B*1024*/)
{
    int i = blockIdx.x * blockDim.x + threadIdx.x;
    if (i >= total) return;
    int b = i >> 10; int pix = i & 1023;
    float z[5];
    #pragma unroll
    for (int c = 0; c < 5; ++c) z[c] = ze[((b * 5 + c) << 10) + pix];
    int best = 0; float bd = 3.4e38f;
    for (int k = 0; k < 512; ++k) {
        float d = 0.f;
        #pragma unroll
        for (int c = 0; c < 5; ++c) { float t = z[c] - cb[k * 5 + c]; d += t * t; }
        if (d < bd) { bd = d; best = k; }   // first-min tie-break == argmin
    }
    #pragma unroll
    for (int c = 0; c < 5; ++c) zq[((b * 5 + c) << 10) + pix] = cb[best * 5 + c];
}

__global__ void tanh_k(float* __restrict__ x, int total)
{
    int i = blockIdx.x * blockDim.x + threadIdx.x;
    if (i < total) x[i] = tanhf(x[i]);
}

// ---------------------------------------------------------------------------
// Input index map (setup_inputs depth-first dict insertion order):
//  0:x | enc.rb1: 1 c1.w 2 c1.b 3 bn1.g 4 bn1.b 5 c2.w 6 c2.b 7 bn2.g 8 bn2.b
//  enc.rb2: 9..16 | enc.rb3: 17..24 | enc.head: 25 w 26 b
//  dec: 27 c0.w 28 c0.b 29 bn1.g 30 bn1.b | rb1: 31..38
//       39 c2.w 40 c2.b 41 bn2.g 42 bn2.b | rb2: 43..50
//       51 c3.w 52 c3.b 53 bn3.g 54 bn3.b | rb3: 55..62
//       63 cf.w 64 cf.b | 65 codebook
// ---------------------------------------------------------------------------
extern "C" void kernel_launch(void* const* d_in, const int* in_sizes, int n_in,
                              void* d_out, int out_size, void* d_ws, size_t ws_size,
                              hipStream_t stream)
{
    const float* X = (const float*)d_in[0];
    const int Bn = 8;

    float* ws   = (float*)d_ws;
    const long long OFF = 33554432LL;      // 8*64*256*256 floats (max live tensor)
    float* A     = ws;
    float* Bf    = ws + OFF;
    float* stats = ws + 2 * OFF;           // 2*256 floats

    float* out = (float*)d_out;
    const int OUT_IMG = 8 * 3 * 256 * 256;     // 1572864
    const int ZE_SZ   = 8 * 5 * 32 * 32;       // 40960
    float* ze = out + OUT_IMG;
    float* zq = ze + ZE_SZ;

    auto conv = [&](const float* in, int wi, float* o, int Cin, int H, int W,
                    int Cout, int up) {
        const float* wp = (const float*)d_in[wi];
        const float* bp = (const float*)d_in[wi + 1];
        int lw = __builtin_ctz(W);
        unsigned gx = (unsigned)((Bn * H * W) / 128);
        if (Cout >= 128) {
            dim3 g(gx, (unsigned)(Cout / 128));
            if (up) conv3x3_wmma<1,128><<<g, 256, 0, stream>>>(in, wp, bp, o, Cin, H, W, Cout, lw);
            else    conv3x3_wmma<0,128><<<g, 256, 0, stream>>>(in, wp, bp, o, Cin, H, W, Cout, lw);
        } else {
            dim3 g(gx, (unsigned)((Cout + 63) / 64));
            if (up) conv3x3_wmma<1,64><<<g, 256, 0, stream>>>(in, wp, bp, o, Cin, H, W, Cout, lw);
            else    conv3x3_wmma<0,64><<<g, 256, 0, stream>>>(in, wp, bp, o, Cin, H, W, Cout, lw);
        }
    };
    auto bn = [&](float* x, int gi, int C, int H, int W) {
        int HW = H * W;
        bn_stats_k<<<C, 256, 0, stream>>>(x, stats, C, HW, Bn);
        int total = Bn * C * HW;
        bn_lrelu_k<<<(total + 255) / 256, 256, 0, stream>>>(
            x, stats, (const float*)d_in[gi], (const float*)d_in[gi + 1],
            C - 1, __builtin_ctz(HW), total);
    };
    auto pool = [&](const float* in, float* o, int C, int H, int W) {
        int total = Bn * C * (H / 2) * (W / 2);
        maxpool2_k<<<(total + 255) / 256, 256, 0, stream>>>(in, o, total,
                                                            __builtin_ctz(W));
    };

    // ---------------- Encoder ----------------
    conv(X, 1, A, 3, 256, 256, 64, 0);   bn(A, 3, 64, 256, 256);
    conv(A, 5, Bf, 64, 256, 256, 64, 0); bn(Bf, 7, 64, 256, 256);
    pool(Bf, A, 64, 256, 256);                               // [8,64,128,128]
    conv(A, 9, Bf, 64, 128, 128, 128, 0);   bn(Bf, 11, 128, 128, 128);
    conv(Bf, 13, A, 128, 128, 128, 128, 0); bn(A, 15, 128, 128, 128);
    pool(A, Bf, 128, 128, 128);                              // [8,128,64,64]
    conv(Bf, 17, A, 128, 64, 64, 256, 0);   bn(A, 19, 256, 64, 64);
    conv(A, 21, Bf, 256, 64, 64, 256, 0);   bn(Bf, 23, 256, 64, 64);
    pool(Bf, A, 256, 64, 64);                                // [8,256,32,32]

    // head 1x1 -> z_e (directly into d_out region)
    {
        int total = Bn * 5 * 1024;
        conv1x1_k<<<(total + 255) / 256, 256, 0, stream>>>(
            A, (const float*)d_in[25], (const float*)d_in[26], ze, 256, 5, total);
    }
    // vector quantize -> z_q (== z_q_ste numerically in forward)
    {
        int total = Bn * 1024;
        vq_k<<<(total + 255) / 256, 256, 0, stream>>>(
            ze, (const float*)d_in[65], zq, total);
    }

    // ---------------- Decoder (upsample fused into conv im2col) ----------------
    conv(zq, 27, A, 5, 32, 32, 256, 0);      bn(A, 29, 256, 32, 32);
    conv(A, 31, Bf, 256, 64, 64, 256, 1);    bn(Bf, 33, 256, 64, 64);     // up+rb1.c1
    conv(Bf, 35, A, 256, 64, 64, 256, 0);    bn(A, 37, 256, 64, 64);      // rb1.c2
    conv(A, 39, Bf, 256, 128, 128, 128, 1);  bn(Bf, 41, 128, 128, 128);   // up+c2
    conv(Bf, 43, A, 128, 128, 128, 128, 0);  bn(A, 45, 128, 128, 128);    // rb2.c1
    conv(A, 47, Bf, 128, 128, 128, 128, 0);  bn(Bf, 49, 128, 128, 128);   // rb2.c2
    conv(Bf, 51, A, 128, 256, 256, 64, 1);   bn(A, 53, 64, 256, 256);     // up+c3
    conv(A, 55, Bf, 64, 256, 256, 64, 0);    bn(Bf, 57, 64, 256, 256);    // rb3.c1
    conv(Bf, 59, A, 64, 256, 256, 64, 0);    bn(A, 61, 64, 256, 256);     // rb3.c2
    conv(A, 63, out, 64, 256, 256, 3, 0);                                 // cf
    tanh_k<<<(OUT_IMG + 255) / 256, 256, 0, stream>>>(out, OUT_IMG);
}
